// GaussianUpsampling_23424751633094
// MI455X (gfx1250) — matrix operand order repaired and checked by reading, not compiled, not run on hardware
//
#include <hip/hip_runtime.h>
#include <hip/hip_bf16.h>
#include <math.h>
#include <stdint.h>

// ---------- problem constants (from setup_inputs) ----------
#define B_   32
#define N_   256
#define T_   2048
#define H_   576
#define P_   32
#define C1_  32
#define C2_  64
#define D_   640    // H_ + C2_
#define G_   320    // D_/2
#define G3_  960    // 3*G_
#define OW_  608    // H_ + P_
#define BN_EPS 1e-5f
#define MASK_SCORE (-1.0e15f)

#define BM_ 64
#define BN2_ 64
#define BK_ 32

typedef __attribute__((ext_vector_type(16))) __bf16 v16bf;
typedef __attribute__((ext_vector_type(8)))  float  v8f;

__device__ __forceinline__ float gelu_exact(float x) {
    return 0.5f * x * (1.0f + erff(x * 0.70710678118654752f));
}
__device__ __forceinline__ float sigm(float x) {
    return 1.0f / (1.0f + __expf(-x));
}
// K-index inside a 16x32 bf16 A-fragment for element e of lane with hi=lane>>4
__device__ __forceinline__ int a_frag_k(int e, int hi) {
    int v = e >> 1, j = e & 1;
    return 2 * v + j + ((v >= 4) ? 8 : 0) + 8 * hi;
}
// CDNA5 async global->LDS copy of 16 bytes per lane (ASYNCcnt-tracked)
__device__ __forceinline__ void async_cp16(uint32_t lds_byte_off, const void* gptr) {
    asm volatile("global_load_async_to_lds_b128 %0, %1, off"
                 :: "v"(lds_byte_off), "v"(gptr) : "memory");
}
__device__ __forceinline__ void async_wait0() {
    asm volatile("s_wait_asynccnt 0x0" ::: "memory");
}
__device__ __forceinline__ uint32_t lds_off(const void* p) {
    return (uint32_t)(uintptr_t)p;   // LDS aperture: low 32 bits = LDS byte offset
}

// ---------- small kernels ----------
__global__ void k_cumsum(const float* __restrict__ dur, float* __restrict__ cbuf) {
    int b = threadIdx.x;
    if (b >= B_) return;
    float run = 0.f;
    for (int n = 0; n < N_; ++n) {
        float d = dur[b * N_ + n];
        run += d;
        cbuf[b * N_ + n] = run - 0.5f * d;
    }
}

__global__ __launch_bounds__(256) void k_conv1(
    const float* __restrict__ dur, const float* __restrict__ w1, const float* __restrict__ b1,
    const float* __restrict__ g1, const float* __restrict__ be1,
    const float* __restrict__ mu1, const float* __restrict__ va1,
    float* __restrict__ pd1)
{
    int b = blockIdx.x, n = threadIdx.x;
    __shared__ float ld[N_ + 2];
    if (n == 0) { ld[0] = 0.f; ld[N_ + 1] = 0.f; }
    ld[n + 1] = dur[b * N_ + n];
    __syncthreads();
    for (int c = 0; c < C1_; ++c) {
        float y = w1[c*3+0]*ld[n] + w1[c*3+1]*ld[n+1] + w1[c*3+2]*ld[n+2] + b1[c];
        float inv = rsqrtf(va1[c] + BN_EPS);
        y = (y - mu1[c]) * (inv * g1[c]) + be1[c];
        pd1[((size_t)b * C1_ + c) * N_ + n] = gelu_exact(y);
    }
}

__global__ __launch_bounds__(256) void k_conv2(
    const float* __restrict__ pd1, const float* __restrict__ w2, const float* __restrict__ b2,
    const float* __restrict__ g2, const float* __restrict__ be2,
    const float* __restrict__ mu2, const float* __restrict__ va2,
    __bf16* __restrict__ ginb)
{
    int b = blockIdx.x, n = threadIdx.x;
    __shared__ float ld[C1_][N_ + 2];
    for (int c = 0; c < C1_; ++c) {
        ld[c][n + 1] = pd1[((size_t)b * C1_ + c) * N_ + n];
        if (n == 0) { ld[c][0] = 0.f; ld[c][N_ + 1] = 0.f; }
    }
    __syncthreads();
    for (int c = 0; c < C2_; ++c) {
        float y = b2[c];
        for (int ci = 0; ci < C1_; ++ci) {
            const float* w = &w2[(c * C1_ + ci) * 3];
            y += w[0]*ld[ci][n] + w[1]*ld[ci][n+1] + w[2]*ld[ci][n+2];
        }
        float inv = rsqrtf(va2[c] + BN_EPS);
        y = (y - mu2[c]) * (inv * g2[c]) + be2[c];
        ginb[((size_t)b * N_ + n) * D_ + H_ + c] = (__bf16)gelu_exact(y);
    }
}

// pack encoder rows into bf16 gru input and a bf16 copy for the attention GEMM
__global__ void k_pack_enc(const float* __restrict__ enc,
                           __bf16* __restrict__ ginb, __bf16* __restrict__ encb) {
    size_t idx = (size_t)blockIdx.x * blockDim.x + threadIdx.x;
    if (idx >= (size_t)B_ * N_ * H_) return;
    size_t row = idx / H_;
    int h = (int)(idx % H_);
    __bf16 v = (__bf16)enc[idx];
    ginb[row * D_ + h] = v;
    encb[idx] = v;
}

// w_ih (3G x D) -> bf16 transposed [k][gate] so async tile rows are contiguous
__global__ void k_prep_wih(const float* __restrict__ wih, __bf16* __restrict__ wT) {
    int idx = blockIdx.x * blockDim.x + threadIdx.x;
    if (idx >= D_ * G3_) return;
    int g = idx / D_, k = idx % D_;
    wT[(size_t)k * G3_ + g] = (__bf16)wih[idx];
}

__global__ void k_transpose_whh(const float* __restrict__ whh, float* __restrict__ whhT) {
    int idx = blockIdx.x * blockDim.x + threadIdx.x;
    if (idx >= G_ * G3_) return;
    int g = idx / G_, k = idx % G_;
    whhT[k * G3_ + g] = whh[idx];
}

// ---------- gx = gru_in x w_ih^T + b_ih : 64x64 block tile, async double-buffer, bf16 WMMA ----------
__global__ __launch_bounds__(128) void k_gemm_gx(
    const __bf16* __restrict__ ginb,
    const __bf16* __restrict__ wT_f, const __bf16* __restrict__ wT_b,
    const float* __restrict__ bih_f, const float* __restrict__ bih_b,
    float* __restrict__ gxf, float* __restrict__ gxb)
{
    const int dir = blockIdx.z;
    const __bf16* wT = dir ? wT_b : wT_f;
    const float* bih = dir ? bih_b : bih_f;
    float* gx = dir ? gxb : gxf;
    const int mbase = blockIdx.x * BM_;
    const int nbase = blockIdx.y * BN2_;
    const int tid = threadIdx.x;
    const int wvid = tid >> 5, lane = tid & 31;
    const int hi = lane >> 4, lr = lane & 15;

    __shared__ __bf16 Ab[2][BM_ * BK_];   // [row][k], rows of 64B
    __shared__ __bf16 Bt[2][BK_ * BN2_];  // [k][gate], rows of 128B

    auto issue = [&](int kstep, int nb) {
        int k0 = kstep * BK_;
        uint32_t abase = lds_off(&Ab[nb][0]);
        uint32_t bbase = lds_off(&Bt[nb][0]);
        #pragma unroll
        for (int i = 0; i < 2; ++i) {          // A: 64x32 bf16 = 256 x 16B chunks
            int cid = i * 128 + tid;
            int row = cid >> 2, seg = cid & 3;
            async_cp16(abase + cid * 16,
                       (const char*)ginb + ((size_t)(mbase + row) * D_ + k0) * 2 + seg * 16);
        }
        #pragma unroll
        for (int i = 0; i < 2; ++i) {          // B: 32x64 bf16 = 256 x 16B chunks
            int cid = i * 128 + tid;
            int kk = cid >> 3, seg = cid & 7;
            async_cp16(bbase + cid * 16,
                       (const char*)wT + ((size_t)(k0 + kk) * G3_ + nbase) * 2 + seg * 16);
        }
    };

    v8f acc[4] = {};
    const int KS = D_ / BK_;  // 20
    issue(0, 0);
    for (int k = 0; k < KS; ++k) {
        int nb = k & 1;
        async_wait0();
        __syncthreads();
        if (k + 1 < KS) issue(k + 1, nb ^ 1);
        v16bf af, bfr;
        #pragma unroll
        for (int e = 0; e < 16; ++e)
            af[e] = Ab[nb][(wvid * 16 + lr) * BK_ + a_frag_k(e, hi)];
        int kb = lr + 16 * hi;
        #pragma unroll
        for (int ns = 0; ns < 4; ++ns) {
            #pragma unroll
            for (int e = 0; e < 16; ++e)
                bfr[e] = Bt[nb][a_frag_k(e, hi) * BN2_ + ns * 16 + lr];
            acc[ns] = __builtin_amdgcn_wmma_f32_16x16x32_bf16(false, af, false, bfr,
                                                              (short)0, acc[ns], false, false);
        }
    }
    #pragma unroll
    for (int ns = 0; ns < 4; ++ns) {
        int col = nbase + ns * 16 + lr;
        float bias = bih[col];
        #pragma unroll
        for (int v = 0; v < 8; ++v) {
            int row = mbase + wvid * 16 + v + 8 * hi;
            gx[(size_t)row * G3_ + col] = acc[ns][v] + bias;
        }
    }
}

// ---------- sequential GRU recurrence: one block per (batch, direction) ----------
__global__ __launch_bounds__(960) void k_gru(
    const float* __restrict__ gxf, const float* __restrict__ gxb,
    const float* __restrict__ whhT_f, const float* __restrict__ whhT_b,
    const float* __restrict__ bhh_f, const float* __restrict__ bhh_b,
    float* __restrict__ hf, float* __restrict__ hb)
{
    const int b = blockIdx.x;
    const int dir = blockIdx.y;
    const float* gx   = dir ? gxb    : gxf;
    const float* whhT = dir ? whhT_b : whhT_f;
    const float* bhh  = dir ? bhh_b  : bhh_f;
    float* hout       = dir ? hb     : hf;
    const int g = threadIdx.x;

    __shared__ float h_lds[G_];
    __shared__ float gh_lds[G3_];

    if (g < G_) h_lds[g] = 0.0f;
    __syncthreads();

    const float bias = bhh[g];
    for (int step = 0; step < N_; ++step) {
        int t = dir ? (N_ - 1 - step) : step;
        float gh = bias;
        #pragma unroll 4
        for (int k = 0; k < G_; ++k)
            gh += whhT[k * G3_ + g] * h_lds[k];   // coalesced across g; h broadcast
        gh_lds[g] = gh;
        __syncthreads();
        float hn = 0.f;
        if (g < G_) {
            size_t base = ((size_t)b * N_ + t) * G3_;
            float xr = gx[base + g];
            float xz = gx[base + G_ + g];
            float xn = gx[base + 2 * G_ + g];
            float r  = sigm(xr + gh_lds[g]);
            float z  = sigm(xz + gh_lds[G_ + g]);
            float nn = tanhf(xn + r * gh_lds[2 * G_ + g]);
            hn = (1.0f - z) * nn + z * h_lds[g];
            hout[((size_t)b * N_ + t) * G_ + g] = hn;
        }
        __syncthreads();
        if (g < G_) h_lds[g] = hn;
        __syncthreads();
    }
}

// ---------- range_params = softplus([h_f, h_b] . range_w), one wave per row ----------
__global__ __launch_bounds__(256) void k_range(
    const float* __restrict__ hf, const float* __restrict__ hb,
    const float* __restrict__ rw, float* __restrict__ range)
{
    int row  = (blockIdx.x * blockDim.x + threadIdx.x) >> 5;
    int lane = threadIdx.x & 31;
    if (row >= B_ * N_) return;
    float acc = 0.f;
    for (int d = lane; d < G_; d += 32) acc += hf[(size_t)row * G_ + d] * rw[d];
    for (int d = lane; d < G_; d += 32) acc += hb[(size_t)row * G_ + d] * rw[G_ + d];
    #pragma unroll
    for (int off = 16; off; off >>= 1) acc += __shfl_xor(acc, off, 32);
    if (lane == 0)
        range[row] = (acc > 20.f) ? acc : log1pf(__expf(acc));
}

// ---------- per-(b,t) online softmax stats over n ----------
__global__ void k_mstats(const float* __restrict__ cbuf, const float* __restrict__ rbuf,
                         const int* __restrict__ lens,
                         float* __restrict__ mrow, float* __restrict__ srow)
{
    int idx = blockIdx.x * blockDim.x + threadIdx.x;
    if (idx >= B_ * T_) return;
    int b = idx / T_, t = idx % T_;
    int len = lens[b];
    float tf = (float)t;
    float m = -3.0e38f, s = 0.f;
    for (int n = 0; n < N_; ++n) {
        float w;
        if (n < len) {
            float diff = (tf - cbuf[b * N_ + n]) / rbuf[b * N_ + n];
            w = -diff * diff;
        } else {
            w = MASK_SCORE;
        }
        if (w > m) { s = s * __expf(m - w) + 1.0f; m = w; }
        else       { s += __expf(w - m); }
    }
    mrow[idx] = m;
    srow[idx] = s;
}

// ---------- attention: up[b,t,h] = sum_n softmax(w)[b,n,t]*enc[b,n,h] ----------
// 64x64 block tile; probs generated on the fly into LDS; enc async-copied; bf16 WMMA
__global__ __launch_bounds__(128) void k_attn(
    const __bf16* __restrict__ encb,
    const float* __restrict__ cbuf, const float* __restrict__ rbuf,
    const float* __restrict__ mrow, const float* __restrict__ srow,
    const int* __restrict__ lens, float* __restrict__ out)
{
    const int b = blockIdx.z;
    const int tbase = blockIdx.x * 64;
    const int hbase = blockIdx.y * 64;
    const int tid = threadIdx.x;
    const int wvid = tid >> 5, lane = tid & 31;
    const int hi = lane >> 4, lr = lane & 15;
    const int len = lens[b];

    __shared__ __bf16 Ab[2][64 * BK_];    // probs [t_local][n_local]
    __shared__ __bf16 Bt[2][BK_ * 64];    // enc   [n_local][h_local]

    auto issueB = [&](int kstep, int nb) {
        int k0 = kstep * BK_;
        uint32_t bbase = lds_off(&Bt[nb][0]);
        #pragma unroll
        for (int i = 0; i < 2; ++i) {      // 32x64 bf16 = 256 x 16B chunks
            int cid = i * 128 + tid;
            int kk = cid >> 3, seg = cid & 7;
            async_cp16(bbase + cid * 16,
                       (const char*)encb + ((size_t)(b * N_ + k0 + kk) * H_ + hbase) * 2 + seg * 16);
        }
    };
    auto storeA = [&](int kstep, int nb) {
        int k0 = kstep * BK_;
        int row = tid >> 1;                // each thread: one t row, 16 consecutive n
        int t = tbase + row;
        float mr = mrow[b * T_ + t];
        float tf = (float)t;
        #pragma unroll
        for (int i = 0; i < 16; ++i) {
            int kk = (tid & 1) * 16 + i;
            int n = k0 + kk;
            float w;
            if (n < len) {
                float diff = (tf - cbuf[b * N_ + n]) / rbuf[b * N_ + n];
                w = -diff * diff;
            } else {
                w = MASK_SCORE;
            }
            Ab[nb][row * BK_ + kk] = (__bf16)__expf(w - mr);
        }
    };

    v8f acc[4] = {};
    const int KS = N_ / BK_;  // 8
    issueB(0, 0);
    storeA(0, 0);
    for (int k = 0; k < KS; ++k) {
        int nb = k & 1;
        async_wait0();
        __syncthreads();
        if (k + 1 < KS) { issueB(k + 1, nb ^ 1); storeA(k + 1, nb ^ 1); }
        v16bf af, bfr;
        int kb = lr + 16 * hi;
        #pragma unroll
        for (int e = 0; e < 16; ++e)
            bfr[e] = Bt[nb][a_frag_k(e, hi) * 64 + wvid * 16 + lr];
        #pragma unroll
        for (int ms = 0; ms < 4; ++ms) {
            #pragma unroll
            for (int e = 0; e < 16; ++e)
                af[e] = Ab[nb][(ms * 16 + lr) * BK_ + a_frag_k(e, hi)];
            acc[ms] = __builtin_amdgcn_wmma_f32_16x16x32_bf16(false, af, false, bfr,
                                                              (short)0, acc[ms], false, false);
        }
    }
    int col = hbase + wvid * 16 + lr;
    #pragma unroll
    for (int ms = 0; ms < 4; ++ms) {
        #pragma unroll
        for (int v = 0; v < 8; ++v) {
            int t = tbase + ms * 16 + v + 8 * hi;
            out[((size_t)b * T_ + t) * OW_ + col] = acc[ms][v] / srow[b * T_ + t];
        }
    }
}

__global__ void k_frames(const float* __restrict__ frames, float* __restrict__ out) {
    size_t idx = (size_t)blockIdx.x * blockDim.x + threadIdx.x;
    if (idx >= (size_t)B_ * T_ * P_) return;
    size_t row = idx / P_;
    int p = (int)(idx % P_);
    out[row * OW_ + H_ + p] = frames[idx];
}

// ---------- launcher ----------
extern "C" void kernel_launch(void* const* d_in, const int* in_sizes, int n_in,
                              void* d_out, int out_size, void* d_ws, size_t ws_size,
                              hipStream_t stream) {
    (void)in_sizes; (void)n_in; (void)out_size; (void)ws_size;
    const float* enc    = (const float*)d_in[0];
    const float* dur    = (const float*)d_in[1];
    const float* frames = (const float*)d_in[2];
    const int*   lens   = (const int*)  d_in[3];
    const float* c1w = (const float*)d_in[4];  const float* c1b = (const float*)d_in[5];
    const float* g1  = (const float*)d_in[6];  const float* be1 = (const float*)d_in[7];
    const float* mu1 = (const float*)d_in[8];  const float* va1 = (const float*)d_in[9];
    const float* c2w = (const float*)d_in[10]; const float* c2b = (const float*)d_in[11];
    const float* g2  = (const float*)d_in[12]; const float* be2 = (const float*)d_in[13];
    const float* mu2 = (const float*)d_in[14]; const float* va2 = (const float*)d_in[15];
    const float* wih_f = (const float*)d_in[16]; const float* whh_f = (const float*)d_in[17];
    const float* bih_f = (const float*)d_in[18]; const float* bhh_f = (const float*)d_in[19];
    const float* wih_b = (const float*)d_in[20]; const float* whh_b = (const float*)d_in[21];
    const float* bih_b = (const float*)d_in[22]; const float* bhh_b = (const float*)d_in[23];
    const float* rw    = (const float*)d_in[24];
    float* out = (float*)d_out;

    float* ws     = (float*)d_ws;
    float* cbuf   = ws;                               // B*N
    float* range  = cbuf   + B_ * N_;                 // B*N
    float* pd1    = range  + B_ * N_;                 // B*C1*N
    float* gxf    = pd1    + (size_t)B_ * C1_ * N_;   // B*N*3G
    float* gxb    = gxf    + (size_t)B_ * N_ * G3_;   // B*N*3G
    float* hf     = gxb    + (size_t)B_ * N_ * G3_;   // B*N*G
    float* hb     = hf     + (size_t)B_ * N_ * G_;    // B*N*G
    float* whhTf  = hb     + (size_t)B_ * N_ * G_;    // G*3G
    float* whhTb  = whhTf  + (size_t)G_ * G3_;        // G*3G
    float* mrow   = whhTb  + (size_t)G_ * G3_;        // B*T
    float* srow   = mrow   + (size_t)B_ * T_;         // B*T
    __bf16* ginb  = (__bf16*)(srow + (size_t)B_ * T_);// B*N*D bf16
    __bf16* encb  = ginb + (size_t)B_ * N_ * D_;      // B*N*H bf16
    __bf16* wTf   = encb + (size_t)B_ * N_ * H_;      // D*3G bf16
    __bf16* wTb   = wTf  + (size_t)D_ * G3_;          // D*3G bf16

    k_cumsum<<<1, 32, 0, stream>>>(dur, cbuf);
    k_conv1<<<B_, N_, 0, stream>>>(dur, c1w, c1b, g1, be1, mu1, va1, pd1);
    k_conv2<<<B_, N_, 0, stream>>>(pd1, c2w, c2b, g2, be2, mu2, va2, ginb);
    {
        size_t total = (size_t)B_ * N_ * H_;
        k_pack_enc<<<(unsigned)((total + 255) / 256), 256, 0, stream>>>(enc, ginb, encb);
    }
    {
        int total = D_ * G3_;
        k_prep_wih<<<(total + 255) / 256, 256, 0, stream>>>(wih_f, wTf);
        k_prep_wih<<<(total + 255) / 256, 256, 0, stream>>>(wih_b, wTb);
    }
    k_gemm_gx<<<dim3((B_ * N_) / BM_, G3_ / BN2_, 2), 128, 0, stream>>>(
        ginb, wTf, wTb, bih_f, bih_b, gxf, gxb);
    {
        int total = G_ * G3_;
        k_transpose_whh<<<(total + 255) / 256, 256, 0, stream>>>(whh_f, whhTf);
        k_transpose_whh<<<(total + 255) / 256, 256, 0, stream>>>(whh_b, whhTb);
    }
    k_gru<<<dim3(B_, 2), G3_, 0, stream>>>(gxf, gxb, whhTf, whhTb, bhh_f, bhh_b, hf, hb);
    k_range<<<(B_ * N_) / 8, 256, 0, stream>>>(hf, hb, rw, range);
    k_mstats<<<(B_ * T_) / 256, 256, 0, stream>>>(cbuf, range, lens, mrow, srow);
    k_attn<<<dim3(T_ / 64, H_ / 64, B_), 128, 0, stream>>>(
        encb, cbuf, range, mrow, srow, lens, out);
    {
        size_t total = (size_t)B_ * T_ * P_;
        k_frames<<<(unsigned)((total + 255) / 256), 256, 0, stream>>>(frames, out);
    }
}
